// ResidualMap_12163347382556
// MI455X (gfx1250) — matrix operand order repaired and checked by reading, not compiled, not run on hardware
//
#include <hip/hip_runtime.h>
#include <hip/hip_bf16.h>

#define HH 1088
#define WW 1920
#define NN (HH * WW)
#define KTOP 1024
#define BINS 8192
#define CAPT 4096
#define TLOW 0.3f
#define THIGH 0.6f
#define IOU_LOW_T 0.1f
#define IOU_HI_T 0.05f

typedef __attribute__((ext_vector_type(16))) _Float16 v16h;
typedef __attribute__((ext_vector_type(8)))  float    v8f;
typedef unsigned long long u64;
typedef unsigned int u32;

#define AS1 __attribute__((address_space(1)))
#define AS3 __attribute__((address_space(3)))

#if defined(__AMDGCN__) && __has_builtin(__builtin_amdgcn_global_load_async_to_lds_b32) && \
    __has_builtin(__builtin_amdgcn_s_wait_asynccnt)
#define HAVE_ASYNC_LDS 1
#else
#define HAVE_ASYNC_LDS 0
#endif

// ---- ordered-float <-> uint key helpers (monotone for all floats) ----
__device__ __forceinline__ u32 fkey(float s) {
    u32 b = __float_as_uint(s);
    return (b & 0x80000000u) ? ~b : (b | 0x80000000u);
}
__device__ __forceinline__ float fkey_inv(u32 u) {
    u32 b = (u & 0x80000000u) ? (u & 0x7FFFFFFFu) : ~u;
    return __uint_as_float(b);
}

__device__ __forceinline__ bool iou_gt(float4 A, float4 B, float th) {
    float iw = fminf(A.z, B.z) - fmaxf(A.x, B.x); iw = fmaxf(iw, 0.0f);
    float ih = fminf(A.w, B.w) - fmaxf(A.y, B.y); ih = fmaxf(ih, 0.0f);
    float inter = iw * ih;
    float areaA = (A.z - A.x) * (A.w - A.y);
    float areaB = (B.z - B.x) * (B.w - B.y);
    return inter / (areaA + areaB - inter) > th;
}

// In-LDS bitonic sort, descending, n = power of two.
__device__ void bitonic_desc(u64* key, int n, int t, int nt) {
    for (int kk = 2; kk <= n; kk <<= 1) {
        for (int j = kk >> 1; j > 0; j >>= 1) {
            __syncthreads();
            for (int i = t; i < n; i += nt) {
                int ixj = i ^ j;
                if (ixj > i) {
                    u64 x = key[i], y = key[ixj];
                    bool firstHalf = ((i & kk) == 0);
                    if (firstHalf ? (x < y) : (x > y)) { key[i] = y; key[ixj] = x; }
                }
            }
        }
    }
    __syncthreads();
}

// ---------------- pipeline kernels ----------------

__global__ void init_kernel(int* hist, int* scal, int* cnts) {
    int i = blockIdx.x * blockDim.x + threadIdx.x;
    if (i < 2 * BINS) hist[i] = 0;
    if (i < 8) cnts[i] = 0;
    if (i == 0) { scal[0] = 0x7F7FFFFF; scal[1] = 0; scal[2] = 0; scal[3] = 0; }
}

// r = 0.5*(sum_c|x1-x2| + sum_c|x2-x3|), plus global min/max (nonneg -> int-bit atomics ok)
__global__ void residual_kernel(const float* __restrict__ x1, const float* __restrict__ x2,
                                const float* __restrict__ x3, float* __restrict__ r,
                                int* __restrict__ scal) {
    int i = blockIdx.x * blockDim.x + threadIdx.x;
    float v = 0.0f;
    if (i < NN) {
        // gfx1250 prefetch path (global_prefetch_b8): pull next channel planes toward L2
        __builtin_prefetch(x1 + i + NN, 0, 1);
        __builtin_prefetch(x2 + i + NN, 0, 1);
        __builtin_prefetch(x3 + i + NN, 0, 1);
        float a = fabsf(x1[i] - x2[i]) + fabsf(x1[i + NN] - x2[i + NN]) +
                  fabsf(x1[i + 2 * NN] - x2[i + 2 * NN]);
        float b = fabsf(x2[i] - x3[i]) + fabsf(x2[i + NN] - x3[i + NN]) +
                  fabsf(x2[i + 2 * NN] - x3[i + 2 * NN]);
        v = 0.5f * (a + b);
        r[i] = v;
    }
    __shared__ float smin[256], smax[256];
    int t = threadIdx.x;
    smin[t] = (i < NN) ? v : 3.4e38f;
    smax[t] = (i < NN) ? v : 0.0f;
    __syncthreads();
    for (int s = 128; s > 0; s >>= 1) {
        if (t < s) {
            smin[t] = fminf(smin[t], smin[t + s]);
            smax[t] = fmaxf(smax[t], smax[t + s]);
        }
        __syncthreads();
    }
    if (t == 0) {
        atomicMin(&scal[0], __float_as_int(smin[0]));
        atomicMax(&scal[1], __float_as_int(smax[0]));
    }
}

__global__ void normalize_kernel(const float* __restrict__ r, float* __restrict__ outmap,
                                 const int* __restrict__ scal) {
    int i = blockIdx.x * blockDim.x + threadIdx.x;
    if (i >= NN) return;
    float rmin = __int_as_float(scal[0]);
    float rmax = __int_as_float(scal[1]);
    outmap[i] = (r[i] - rmin) / (rmax - rmin + 1e-6f);
}

// Horizontal 5-tap [1,2,2,2,1]/8 blur (clamp padding) as a banded WMMA GEMM:
//   D(16x16) = A(16x32: input rows, 20 used cols, f16) x B(32x16: band weights, f16)
// The reference's dropped-tap edge correction is folded INTO B:
//   col0==0 tile:     B[2][0]  -= 0.25   (image col 0 is A-column k=2)
//   col0==W-16 tile:  B[17][15] -= 0.25  (image col W-1 is A-column k=17)
// so the store loop is branch-free.
__global__ void blur_w_wmma(const float* __restrict__ in, float* __restrict__ out) {
    const float wt[5] = {0.125f, 0.25f, 0.25f, 0.25f, 0.125f};
    int gtid = blockIdx.x * blockDim.x + threadIdx.x;
    int wave = gtid >> 5;
    int lane = threadIdx.x & 31;
    int tx = wave % (WW / 16);
    int ty = wave / (WW / 16);
    int col0 = tx * 16, row0 = ty * 16;
    int half = lane >> 4, lm = lane & 15;
    const float* rowp = in + (row0 + lm) * WW;
    v16h a, b;
#pragma unroll
    for (int e = 0; e < 16; ++e) {
        // A (16-bit A-matrix 16x32 layout): lane half selects K-block, pairs packed
        int Ka = 8 * half + e + ((e >= 8) ? 8 : 0);
        float av = 0.0f;
        if (Ka < 20) {
            int c = col0 + Ka - 2;
            c = c < 0 ? 0 : (c > WW - 1 ? WW - 1 : c);
            av = rowp[c];
        }
        a[e] = (_Float16)av;
        // B (32x16): same K order as A (lane half selects K-block); N = lane&15
        int Kb = 8 * half + e + ((e >= 8) ? 8 : 0);
        int d = Kb - lm;
        float bv = (Kb < 20 && d >= 0 && d <= 4) ? wt[d] : 0.0f;
        if (col0 == 0 && lm == 0 && Kb == 2) bv -= 0.25f;
        if (col0 == WW - 16 && lm == 15 && Kb == 17) bv -= 0.25f;
        b[e] = (_Float16)bv;
    }
    v8f acc = {};
    acc = __builtin_amdgcn_wmma_f32_16x16x32_f16(false, a, false, b, (short)0, acc, false, false);
    int c = col0 + lm;
#pragma unroll
    for (int d = 0; d < 8; ++d) {
        int rr = row0 + d + 8 * half;
        out[rr * WW + c] = acc[d];
    }
}

// Vertical pass: A = band weights (edge correction folded into A), B = input columns
// staged block-wide through LDS via GLOBAL_LOAD_ASYNC_TO_LDS (ASYNCcnt) when available.
// Block = 8 adjacent tiles in one tile-row: stages a 20 x 128 slab (10 KB LDS).
__global__ void blur_h_wmma(const float* __restrict__ in, float* __restrict__ out,
                            int* __restrict__ scal) {
    const float wt[5] = {0.125f, 0.25f, 0.25f, 0.25f, 0.125f};
    __shared__ float slab[20 * 128];
    int t = threadIdx.x;
    int waveInBlk = t >> 5;
    int lane = t & 31;
    int ty = blockIdx.x / (WW / 16 / 8);          // tile row (8 tiles per block, 15 blocks/row)
    int txbase = (blockIdx.x % (WW / 16 / 8)) * 8;
    int row0 = ty * 16;
    int colbase = txbase * 16;                    // 128-wide column slab
    // --- stage in[row0-2 .. row0+17][colbase .. colbase+127] into LDS (rows clamped) ---
    for (int i = t; i < 20 * 128; i += 256) {
        int lr = i >> 7, lc = i & 127;
        int gr = row0 - 2 + lr;
        gr = gr < 0 ? 0 : (gr > HH - 1 ? HH - 1 : gr);
        const float* src = in + gr * WW + colbase + lc;
#if HAVE_ASYNC_LDS
        __builtin_amdgcn_global_load_async_to_lds_b32((AS1 int*)src, (AS3 int*)&slab[i], 0, 0);
#else
        slab[i] = *src;
#endif
    }
#if HAVE_ASYNC_LDS
    __builtin_amdgcn_s_wait_asynccnt(0);
#endif
    __syncthreads();

    int half = lane >> 4, lm = lane & 15;
    int lcol = waveInBlk * 16 + lm;               // column within the slab
    int col = colbase + lcol;
    v16h a, b;
#pragma unroll
    for (int e = 0; e < 16; ++e) {
        int Ka = 8 * half + e + ((e >= 8) ? 8 : 0);
        int d = Ka - lm;                          // A[m,k] = wt[k-m], m = lane&15
        float av = (d >= 0 && d <= 4) ? wt[d] : 0.0f;
        if (row0 == 0 && lm == 0 && Ka == 2) av -= 0.25f;          // image row 0
        if (row0 == HH - 16 && lm == 15 && Ka == 17) av -= 0.25f;  // image row H-1
        a[e] = (_Float16)av;
        int Kb = 8 * half + e + ((e >= 8) ? 8 : 0); // B[k,n] = slab[k][lcol]
        float bv = (Kb < 20) ? slab[Kb * 128 + lcol] : 0.0f;
        b[e] = (_Float16)bv;
    }
    v8f acc = {};
    acc = __builtin_amdgcn_wmma_f32_16x16x32_f16(false, a, false, b, (short)0, acc, false, false);
    float lmax = 0.0f;
#pragma unroll
    for (int d = 0; d < 8; ++d) {
        int rr = row0 + d + 8 * half;
        float v = acc[d];
        out[rr * WW + col] = v;
        lmax = fmaxf(lmax, v);
    }
    atomicMax(&scal[2], __float_as_int(lmax));
}

// Per-branch score histogram (branch = blockIdx.y; 0 = hi, 1 = lo)
__global__ void hist_kernel(const float* __restrict__ logits, const int* __restrict__ scal,
                            int* __restrict__ hist) {
    __shared__ int sh[BINS];
    int t = threadIdx.x;
    for (int i = t; i < BINS; i += blockDim.x) sh[i] = 0;
    __syncthreads();
    int br = blockIdx.y;
    float inv = 1.0f / (__int_as_float(scal[2]) + 1e-6f);
    for (int i = blockIdx.x * blockDim.x + t; i < NN; i += gridDim.x * blockDim.x) {
        float s = logits[i] * inv;
        bool m = (br == 0) ? (s > THIGH) : (s > TLOW && s < THIGH);
        if (m) {
            int bin = (int)(s * (float)BINS);
            bin = bin < 0 ? 0 : (bin > BINS - 1 ? BINS - 1 : bin);
            atomicAdd(&sh[bin], 1);
        }
    }
    __syncthreads();
    for (int i = t; i < BINS; i += blockDim.x)
        if (sh[i]) atomicAdd(&hist[br * BINS + i], sh[i]);
}

// Find threshold bin T (all candidates in bins > T selected) and #needed from bin T.
__global__ void thresh_kernel(const int* __restrict__ hist, int* __restrict__ cnts) {
    int b = threadIdx.x;
    if (b >= 2) return;
    int above = 0, T = -1, need = 0;
    for (int bin = BINS - 1; bin >= 0; --bin) {
        int h = hist[b * BINS + bin];
        if (above + h > KTOP) { T = bin; need = KTOP - above; break; }
        above += h;
    }
    cnts[b * 4 + 2] = T;
    cnts[b * 4 + 3] = need;
}

// Gather selected (bin > T) and boundary-bin (bin == T) candidates as 64-bit keys.
__global__ void gather_kernel(const float* __restrict__ logits, const int* __restrict__ scal,
                              int* __restrict__ cnts, u64* __restrict__ sel,
                              u64* __restrict__ bint) {
    int i = blockIdx.x * blockDim.x + threadIdx.x;
    if (i >= NN) return;
    float inv = 1.0f / (__int_as_float(scal[2]) + 1e-6f);
    float s = logits[i] * inv;
    if (!(s > TLOW) || s == THIGH) return;
    int br = (s > THIGH) ? 0 : 1;
    int bin = (int)(s * (float)BINS);
    bin = bin < 0 ? 0 : (bin > BINS - 1 ? BINS - 1 : bin);
    int T = cnts[br * 4 + 2];
    u64 key = ((u64)fkey(s) << 32) | (u32)(~(u32)i);   // value desc, index asc
    if (bin > T) {
        int p = atomicAdd(&cnts[br * 4 + 0], 1);
        if (p < KTOP) sel[br * KTOP + p] = key;
    } else if (bin == T) {
        int p = atomicAdd(&cnts[br * 4 + 1], 1);
        if (p < CAPT) bint[br * CAPT + p] = key;
    }
}

// Per-branch: sort boundary bin, assemble exact top-K, sort K, greedy NMS, emit rows.
__global__ __launch_bounds__(1024) void branch_kernel(const int* __restrict__ cnts,
                                                      const u64* __restrict__ sel,
                                                      const u64* __restrict__ bint,
                                                      float* __restrict__ cat) {
    __shared__ u64 skey[CAPT];
    __shared__ u64 fin[KTOP];
    __shared__ float4 sbox[KTOP];
    __shared__ int skeep[KTOP];
    int b = blockIdx.x;
    int t = threadIdx.x;
    int cntA = cnts[b * 4 + 0]; if (cntA > KTOP) cntA = KTOP;
    int cntT = cnts[b * 4 + 1]; if (cntT > CAPT) cntT = CAPT;
    int need = cnts[b * 4 + 3]; if (need > cntT) need = cntT; if (need < 0) need = 0;
    for (int i = t; i < CAPT; i += 1024) skey[i] = (i < cntT) ? bint[b * CAPT + i] : 0ull;
    bitonic_desc(skey, CAPT, t, 1024);
    {
        u64 v = 0ull;
        if (t < cntA) v = sel[b * KTOP + t];
        else if (t < cntA + need) v = skey[t - cntA];
        fin[t] = v;
    }
    bitonic_desc(fin, KTOP, t, 1024);
    u64 key = fin[t];
    float sc = -1.0f; int idx = 0;
    if (key) { sc = fkey_inv((u32)(key >> 32)); idx = (int)(~(u32)key); }
    int xi = idx % WW, yi = idx / WW;
    float4 bx = make_float4((float)(xi - 2), (float)(yi - 2), (float)(xi + 2), (float)(yi + 2));
    sbox[t] = bx;
    skeep[t] = (sc >= 0.0f) ? 1 : 0;
    __syncthreads();
    float th = (b == 0) ? IOU_HI_T : IOU_LOW_T;
    for (int i2 = 0; i2 < KTOP; ++i2) {
        if (skeep[i2] && t > i2 && skeep[t] && iou_gt(sbox[i2], sbox[t], th)) skeep[t] = 0;
        __syncthreads();
    }
    float* row = cat + (b * KTOP + t) * 5;
    row[0] = bx.x; row[1] = bx.y; row[2] = bx.z; row[3] = bx.w;
    row[4] = skeep[t] ? sc : -1.0f;
}

// Concatenate (done by layout), stable sort desc, final NMS, keep-masked write.
__global__ __launch_bounds__(1024) void final_kernel(const float* __restrict__ cat,
                                                     float* __restrict__ outb) {
    __shared__ u64 key2[2048];
    __shared__ float4 box2[2048];
    __shared__ int keep2[2048];
    int t = threadIdx.x;
    for (int i = t; i < 2048; i += 1024) {
        float sc = cat[i * 5 + 4];
        key2[i] = ((u64)fkey(sc) << 32) | (u32)(~(u32)i);   // stable: ties by orig index
    }
    bitonic_desc(key2, 2048, t, 1024);
    for (int i = t; i < 2048; i += 1024) {
        u64 key = key2[i];
        int src = (int)(~(u32)key);
        const float* row = cat + src * 5;
        box2[i] = make_float4(row[0], row[1], row[2], row[3]);
        float sc = fkey_inv((u32)(key >> 32));
        keep2[i] = (sc >= 0.0f) ? 1 : 0;
    }
    __syncthreads();
    for (int i2 = 0; i2 < 2048; ++i2) {
        if (keep2[i2]) {
            for (int j = t; j < 2048; j += 1024)
                if (j > i2 && keep2[j] && iou_gt(box2[i2], box2[j], IOU_HI_T)) keep2[j] = 0;
        }
        __syncthreads();
    }
    for (int i = t; i < 2048; i += 1024) {
        float sc = fkey_inv((u32)(key2[i] >> 32));
        int k = keep2[i];
        float4 bx = box2[i];
        float* o = outb + i * 5;
        o[0] = k ? bx.x : 0.0f;
        o[1] = k ? bx.y : 0.0f;
        o[2] = k ? bx.z : 0.0f;
        o[3] = k ? bx.w : 0.0f;
        o[4] = k ? sc : 0.0f;
    }
}

extern "C" void kernel_launch(void* const* d_in, const int* in_sizes, int n_in,
                              void* d_out, int out_size, void* d_ws, size_t ws_size,
                              hipStream_t stream) {
    (void)in_sizes; (void)n_in; (void)out_size; (void)ws_size;
    const float* x1 = (const float*)d_in[0];
    const float* x2 = (const float*)d_in[1];
    const float* x3 = (const float*)d_in[2];
    float* out = (float*)d_out;

    // workspace layout (all reads initialized per call; deterministic)
    float* buf0 = (float*)d_ws;          // r -> (later) logits
    float* buf1 = buf0 + NN;             // horizontal-blur intermediate
    int*   hist = (int*)(buf1 + NN);     // 2 x BINS
    int*   scal = hist + 2 * BINS;       // min/max/maxlogit bits
    int*   cnts = scal + 8;              // per-branch: above, at, T, need
    u64*   sel  = (u64*)(cnts + 8);      // 2 x KTOP selected-above keys
    u64*   bint = sel + 2 * KTOP;        // 2 x CAPT boundary-bin keys
    float* cat  = (float*)(bint + 2 * CAPT); // 2048 x 5 concat rows

    const int tileBlocks = (HH / 16) * (WW / 16) / 8;  // 8 waves (tiles) per 256-thread block

    init_kernel<<<64, 256, 0, stream>>>(hist, scal, cnts);
    residual_kernel<<<NN / 256, 256, 0, stream>>>(x1, x2, x3, buf0, scal);
    normalize_kernel<<<NN / 256, 256, 0, stream>>>(buf0, out, scal);
    blur_w_wmma<<<tileBlocks, 256, 0, stream>>>(out, buf1);
    blur_h_wmma<<<tileBlocks, 256, 0, stream>>>(buf1, buf0, scal);
    hist_kernel<<<dim3(512, 2), 256, 0, stream>>>(buf0, scal, hist);
    thresh_kernel<<<1, 64, 0, stream>>>(hist, cnts);
    gather_kernel<<<NN / 256, 256, 0, stream>>>(buf0, scal, cnts, sel, bint);
    branch_kernel<<<2, 1024, 0, stream>>>(cnts, sel, bint, cat);
    final_kernel<<<1, 1024, 0, stream>>>(cat, out + NN);
}
